// DSAOp_68324339745458
// MI455X (gfx1250) — compile-verified, matrix operand order checked by reading, not run on hardware
//
#include <hip/hip_runtime.h>

#define T_TOK   512
#define H_HEADS 128
#define NOPE    128
#define ROPE    64
#define KV_LORA 512
#define C_DIM   576                 // KV_LORA + ROPE
#define K_TOP   512
#define VT_LD   72                  // padded k-stride for transposed V tile
#define SCALE   0.041666666666666664f   // (512+64)^-0.5 = 1/24

typedef __attribute__((ext_vector_type(16))) __bf16 v16bf;
typedef __attribute__((ext_vector_type(8)))  float  v8f;
typedef unsigned int uint32;

static __device__ __forceinline__ v8f zero8() {
  v8f z = {0.f, 0.f, 0.f, 0.f, 0.f, 0.f, 0.f, 0.f};
  return z;
}

static __device__ __forceinline__ uint32 pack2_bf16(float a, float b) {
  unsigned short x = __builtin_bit_cast(unsigned short, (__bf16)a);
  unsigned short y = __builtin_bit_cast(unsigned short, (__bf16)b);
  return (uint32)x | ((uint32)y << 16);
}

// A/B fragment from row-major bf16 LDS tile [rows][ld], 16x32 (MxK) per ISA 7.12.2:
// lane L<16  : row L,   K = k0+0..7  and k0+16..23
// lane L>=16 : row L-16, K = k0+8..15 and k0+24..31
static __device__ __forceinline__ v16bf frag_rm_bf16(const __bf16* base, int ld,
                                                     int row0, int k0, int lane) {
  const __bf16* p = base + (row0 + (lane & 15)) * ld + k0 + ((lane >> 4) << 3);
  v16bf f;
#pragma unroll
  for (int i = 0; i < 8; ++i) f[i] = p[i];
#pragma unroll
  for (int i = 0; i < 8; ++i) f[8 + i] = p[16 + i];
  return f;
}

// Same mapping, f32 source converted on load — used for the P matrix.
static __device__ __forceinline__ v16bf frag_rm_f32(const float* base, int ld,
                                                    int row0, int k0, int lane) {
  const float* p = base + (row0 + (lane & 15)) * ld + k0 + ((lane >> 4) << 3);
  v16bf f;
#pragma unroll
  for (int i = 0; i < 8; ++i) f[i] = (__bf16)p[i];
#pragma unroll
  for (int i = 0; i < 8; ++i) f[8 + i] = (__bf16)p[16 + i];
  return f;
}

static __device__ __forceinline__ v8f wmma_bf16(v16bf a, v16bf b, v8f c) {
  return __builtin_amdgcn_wmma_f32_16x16x32_bf16(false, a, false, b, (short)0, c,
                                                 false, false);
}

static __device__ __forceinline__ void wait_asynccnt0() {
#if __has_builtin(__builtin_amdgcn_s_wait_asynccnt)
  __builtin_amdgcn_s_wait_asynccnt(0);
#else
  asm volatile("s_wait_asynccnt 0x0" ::: "memory");
#endif
}

// ---------------------------------------------------------------------------
// Kernel 1: q_lat[t,h,n] = sum_d q[t,h,d]*kbt[h,n,d]  (bf16 out into qc),
// plus rope passthrough qc[t,h,512+r] = q[t,h,128+r].
// grid (H, T/64, 512/128), block 256 (8 waves). C tile 64x128.
// ---------------------------------------------------------------------------
__global__ __launch_bounds__(256, 2) void k_qlat(const float* __restrict__ q,
                                                 const float* __restrict__ kbt,
                                                 __bf16* __restrict__ qc) {
  __shared__ __bf16 As[64 * 128];
  __shared__ __bf16 Bs[128 * 128];
  const int h   = blockIdx.x;
  const int t0  = blockIdx.y * 64;
  const int n0  = blockIdx.z * 128;
  const int tid = threadIdx.x;
  const int lane = tid & 31;
  const int w    = tid >> 5;

  {  // A: q_nope 64x128 f32 -> bf16
    int m = tid >> 2;
    int d0 = (tid & 3) * 32;
    const float* src = q + ((size_t)(t0 + m) * H_HEADS + h) * (NOPE + ROPE) + d0;
    __bf16* dst = As + m * 128 + d0;
#pragma unroll
    for (int i = 0; i < 32; ++i) dst[i] = (__bf16)src[i];
  }
  {  // B: kbt[h, n0.., :] 128x128 f32 -> bf16
    int n = tid >> 1;
    int d0 = (tid & 1) * 64;
    const float* src = kbt + ((size_t)h * KV_LORA + n0 + n) * NOPE + d0;
    __bf16* dst = Bs + n * 128 + d0;
#pragma unroll
    for (int i = 0; i < 64; ++i) dst[i] = (__bf16)src[i];
  }
  if (blockIdx.z == 0) {  // rope passthrough, 64x64 values, 16/thread
    int base = tid * 16;
    int m = base >> 6;
    int r = base & 63;
    const float* src = q + ((size_t)(t0 + m) * H_HEADS + h) * (NOPE + ROPE) + NOPE + r;
    __bf16* dst = qc + ((size_t)(t0 + m) * H_HEADS + h) * C_DIM + KV_LORA + r;
#pragma unroll
    for (int i = 0; i < 16; ++i) dst[i] = (__bf16)src[i];
  }
  __syncthreads();

  const int mt = w >> 1;
  const int nh = w & 1;
  v8f acc[4];
#pragma unroll
  for (int j = 0; j < 4; ++j) acc[j] = zero8();
#pragma unroll
  for (int kc = 0; kc < 4; ++kc) {
    v16bf a = frag_rm_bf16(As, 128, mt * 16, kc * 32, lane);
#pragma unroll
    for (int j = 0; j < 4; ++j) {
      v16bf b = frag_rm_bf16(Bs, 128, (nh * 4 + j) * 16, kc * 32, lane);
      acc[j] = wmma_bf16(a, b, acc[j]);
    }
  }
  const int rowadd = (lane < 16) ? 0 : 8;
  const int col = lane & 15;
#pragma unroll
  for (int j = 0; j < 4; ++j) {
    int n = n0 + (nh * 4 + j) * 16 + col;
#pragma unroll
    for (int r = 0; r < 8; ++r) {
      int m = mt * 16 + r + rowadd;
      qc[((size_t)(t0 + m) * H_HEADS + h) * C_DIM + n] = (__bf16)acc[j][r];
    }
  }
}

// ---------------------------------------------------------------------------
// Kernel 2: attention for one (token, 64-head half). grid (T, 2), block 512.
// Dynamic LDS: Qc 64x576 bf16 | KV 64x576 bf16 (pass2: VT 512x72) |
//              SC 64x512 f32 | RED 64x8 f32
// ---------------------------------------------------------------------------
__global__ __launch_bounds__(512, 1) void k_attn(__bf16* qc,
                                                 const float* __restrict__ kv,
                                                 const int* __restrict__ topk) {
  extern __shared__ char smem_raw[];
  __bf16* Qc  = (__bf16*)smem_raw;            // 64*576
  __bf16* KV  = Qc + 64 * C_DIM;              // 64*576 bf16 (= 512*72 for VT)
  float*  SC  = (float*)(KV + 64 * C_DIM);    // 64*512
  float*  RED = SC + 64 * K_TOP;              // 64*8

  const int t   = blockIdx.x;
  const int h0  = blockIdx.y * 64;
  const int tid = threadIdx.x;
  const int lane = tid & 31;
  const int w    = tid >> 5;                  // 0..15
  const int* tkrow = topk + (size_t)t * K_TOP;

  {  // stage q_concat for 64 heads via async global->LDS copy (no VGPR round trip)
    const char* src = (const char*)(qc + ((size_t)t * H_HEADS + h0) * C_DIM);
    const uint32 lds_base = (uint32)((char*)Qc - smem_raw);  // = 0, kept general
#pragma unroll
    for (int k = 0; k < 9; ++k) {             // 64*576*2 B = 512 thr * 9 * 16 B
      uint32 off = (uint32)(tid + k * 512) * 16u;
      uint32 ldsa = lds_base + off;
      const char* g = src + off;
      asm volatile("global_load_async_to_lds_b128 %0, %1, off"
                   :: "v"(ldsa), "v"(g)
                   : "memory");
    }
    wait_asynccnt0();
  }
  __syncthreads();

  const int ht = w >> 2;                      // 4 head-tiles
  const int kt = w & 3;                       // pass1: k-tile; pass2: c-quarter
  const int rowadd = (lane < 16) ? 0 : 8;
  const int col = lane & 15;

  // ---- pass 1: scores = Qc * KVblk^T ----
  for (int kb = 0; kb < 8; ++kb) {
    {  // gather 64 kv rows (f32 -> bf16), 8 threads per row, row-major
      int r = tid >> 3, j = tid & 7;
      const float* src = kv + (size_t)tkrow[kb * 64 + r] * C_DIM + j * 72;
      __bf16* dst = KV + r * C_DIM + j * 72;
#pragma unroll
      for (int i = 0; i < 72; ++i) dst[i] = (__bf16)src[i];
    }
    __syncthreads();
    v8f acc = zero8();
#pragma unroll
    for (int kc = 0; kc < 18; ++kc) {         // C = 576 = 18 * 32
      v16bf a = frag_rm_bf16(Qc, C_DIM, ht * 16, kc * 32, lane);
      v16bf b = frag_rm_bf16(KV, C_DIM, kt * 16, kc * 32, lane);
      acc = wmma_bf16(a, b, acc);
    }
#pragma unroll
    for (int r = 0; r < 8; ++r)
      SC[(ht * 16 + r + rowadd) * K_TOP + kb * 64 + kt * 16 + col] = acc[r] * SCALE;
    __syncthreads();
  }

  // ---- softmax over K=512, 8 threads per head row ----
  {
    int row = tid >> 3, j = tid & 7;
    float* srow = SC + row * K_TOP + j * 64;
    float m = -3.0e38f;
    for (int i = 0; i < 64; ++i) m = fmaxf(m, srow[i]);
    RED[row * 8 + j] = m;
    __syncthreads();
    float mm = RED[row * 8];
#pragma unroll
    for (int i = 1; i < 8; ++i) mm = fmaxf(mm, RED[row * 8 + i]);
    float s = 0.f;
    for (int i = 0; i < 64; ++i) { float e = __expf(srow[i] - mm); srow[i] = e; s += e; }
    __syncthreads();                           // done reading maxes
    RED[row * 8 + j] = s;
    __syncthreads();
    float tot = RED[row * 8];
#pragma unroll
    for (int i = 1; i < 8; ++i) tot += RED[row * 8 + i];
    float inv = 1.f / tot;
    for (int i = 0; i < 64; ++i) srow[i] *= inv;
  }
  __syncthreads();

  // ---- pass 2: O = P * V.  V gathered TRANSPOSED: VT[c][k], ld = 72 ----
  __bf16* VT = KV;                             // reuse: 512 * 72 bf16 = 72 KB
  const int cq = kt;
  v8f o[8];
#pragma unroll
  for (int j = 0; j < 8; ++j) o[j] = zero8();
  for (int kb = 0; kb < 8; ++kb) {
    {  // 32 row-pairs x 16 c-slots; pack 2 bf16 (k, k+1) per dword store
      int pr = tid >> 4;                       // 0..31
      int j  = tid & 15;                       // c-slot: 32 cols each
      const float* s0 = kv + (size_t)tkrow[kb * 64 + 2 * pr]     * C_DIM + j * 32;
      const float* s1 = kv + (size_t)tkrow[kb * 64 + 2 * pr + 1] * C_DIM + j * 32;
#pragma unroll
      for (int i = 0; i < 32; ++i) {
        int c = j * 32 + i;
        *(uint32*)(VT + c * VT_LD + 2 * pr) = pack2_bf16(s0[i], s1[i]);
      }
    }
    __syncthreads();
#pragma unroll
    for (int kc = 0; kc < 2; ++kc) {
      v16bf a = frag_rm_f32(SC, K_TOP, ht * 16, kb * 64 + kc * 32, lane);
#pragma unroll
      for (int j = 0; j < 8; ++j) {
        // B fragment: lane n = column c, contiguous k run -> 2x b128 LDS loads
        v16bf b = frag_rm_bf16(VT, VT_LD, cq * 128 + j * 16, kc * 32, lane);
        o[j] = wmma_bf16(a, b, o[j]);
      }
    }
    __syncthreads();
  }
  // write attn_lat (bf16) back into qc rows (cols 0..511)
#pragma unroll
  for (int j = 0; j < 8; ++j) {
    int c = cq * 128 + j * 16 + col;
#pragma unroll
    for (int r = 0; r < 8; ++r) {
      int h = h0 + ht * 16 + r + rowadd;
      qc[((size_t)t * H_HEADS + h) * C_DIM + c] = (__bf16)o[j][r];
    }
  }
}

// ---------------------------------------------------------------------------
// Kernel 3: out[t,h,v] = sum_c attn_lat[t,h,c] * vbp[h,v,c].
// grid (H, T/64), block 256 (8 waves). C tile 64x128, K staged 4x128.
// ---------------------------------------------------------------------------
__global__ __launch_bounds__(256, 2) void k_outproj(const __bf16* __restrict__ qc,
                                                    const float* __restrict__ vbp,
                                                    float* __restrict__ out) {
  __shared__ __bf16 As[64 * 128];
  __shared__ __bf16 Bs[128 * 128];
  const int h   = blockIdx.x;
  const int t0  = blockIdx.y * 64;
  const int tid = threadIdx.x;
  const int lane = tid & 31;
  const int w    = tid >> 5;
  const int mt = w >> 1;
  const int nh = w & 1;
  const int rowadd = (lane < 16) ? 0 : 8;
  const int col = lane & 15;
  v8f acc[4];
#pragma unroll
  for (int j = 0; j < 4; ++j) acc[j] = zero8();

  for (int s = 0; s < 4; ++s) {
    int c0 = s * 128;
    {  // A: attn_lat bf16 copy
      int m = tid >> 2;
      int cc = (tid & 3) * 32;
      const uint32* s2 =
          (const uint32*)(qc + ((size_t)(t0 + m) * H_HEADS + h) * C_DIM + c0 + cc);
      uint32* d2 = (uint32*)(As + m * 128 + cc);
#pragma unroll
      for (int i = 0; i < 16; ++i) d2[i] = s2[i];
    }
    {  // B: v_b_proj f32 -> bf16
      int v = tid >> 1;
      int cc = (tid & 1) * 64;
      const float* src = vbp + ((size_t)h * 128 + v) * KV_LORA + c0 + cc;
      __bf16* dst = Bs + v * 128 + cc;
#pragma unroll
      for (int i = 0; i < 64; ++i) dst[i] = (__bf16)src[i];
    }
    __syncthreads();
#pragma unroll
    for (int kc = 0; kc < 4; ++kc) {
      v16bf a = frag_rm_bf16(As, 128, mt * 16, kc * 32, lane);
#pragma unroll
      for (int j = 0; j < 4; ++j) {
        v16bf b = frag_rm_bf16(Bs, 128, (nh * 4 + j) * 16, kc * 32, lane);
        acc[j] = wmma_bf16(a, b, acc[j]);
      }
    }
    __syncthreads();
  }
#pragma unroll
  for (int j = 0; j < 4; ++j) {
    int n = (nh * 4 + j) * 16 + col;
#pragma unroll
    for (int r = 0; r < 8; ++r) {
      int m = mt * 16 + r + rowadd;
      out[((size_t)(t0 + m) * H_HEADS + h) * 128 + n] = acc[j][r];
    }
  }
}

extern "C" void kernel_launch(void* const* d_in, const int* in_sizes, int n_in,
                              void* d_out, int out_size, void* d_ws, size_t ws_size,
                              hipStream_t stream) {
  const float* q   = (const float*)d_in[0];   // [T,H,192]
  const float* kv  = (const float*)d_in[1];   // [S,576]
  const int*   tk  = (const int*)d_in[2];     // [T,512]
  const float* kbt = (const float*)d_in[3];   // [H,512,128]
  const float* vbp = (const float*)d_in[4];   // [H,128,512]
  float* out = (float*)d_out;                 // [T,H,128]
  __bf16* qc = (__bf16*)d_ws;                 // [T,H,576] bf16 = 75.5 MB

  const size_t lds2 = (size_t)(2 * 64 * C_DIM) * sizeof(__bf16)   // Qc + KV/VT
                    + (size_t)(64 * K_TOP) * sizeof(float)        // scores
                    + (size_t)(64 * 8) * sizeof(float);           // reduction
  (void)hipFuncSetAttribute(reinterpret_cast<const void*>(k_attn),
                            hipFuncAttributeMaxDynamicSharedMemorySize, (int)lds2);

  k_qlat<<<dim3(H_HEADS, T_TOK / 64, KV_LORA / 128), 256, 0, stream>>>(q, kbt, qc);
  k_attn<<<dim3(T_TOK, 2), 512, lds2, stream>>>(qc, kv, tk);
  k_outproj<<<dim3(H_HEADS, T_TOK / 64), 256, 0, stream>>>(qc, vbp, out);
}
